// VQLayer_54314156425528
// MI455X (gfx1250) — matrix operand order repaired and checked by reading, not compiled, not run on hardware
//
#include <hip/hip_runtime.h>

typedef __attribute__((ext_vector_type(16))) _Float16 v16h;
typedef __attribute__((ext_vector_type(8)))  float    v8f;

#define D_DIM      256
#define K_DIM      2048
#define N_ROWS     65536          // 64*32*32
#define N_TILES    128            // K_DIM / 16
#define N_CHUNKS   8              // D_DIM / 32
#define TG         4              // code-tiles per LDS slab (32KB)
#define N_GROUPS   (N_TILES / TG)
#define WG_THREADS 256
#define WAVES_WG   8
#define ROWS_WG    128            // 8 waves * 16 rows
#define N_WG       (N_ROWS / ROWS_WG)   // 512
#define ND_TOTAL   16777216.0f
#define SLAB_V16   (TG * N_CHUNKS * 32)       // 1024 v16h per slab
#define SLAB_BYTES (SLAB_V16 * 32)            // 32768 B

// ---------------------------------------------------------------------------
// Prep 1: swizzle codebook into WMMA B-fragment order, f32 -> f16.
// ---------------------------------------------------------------------------
__global__ __launch_bounds__(256) void vq_pack_b(const float* __restrict__ emb,
                                                 _Float16* __restrict__ embP) {
    int t   = blockIdx.x * 256 + threadIdx.x;     // 128*8*32 fragments
    int T   = t >> 8;
    int rem = t & 255;
    int c   = rem >> 5;
    int l   = rem & 31;
    int col = T * 16 + (l & 15);
    int kg  = (l < 16) ? 0 : 16;
    v16h v;
#pragma unroll
    for (int e = 0; e < 16; ++e)
        v[e] = (_Float16)emb[(size_t)(32 * c + kg + e) * K_DIM + col];
    *((v16h*)embP + t) = v;
}

// Prep 2: e2[k] = sum_d emb[d][k]^2 (fp32)
__global__ __launch_bounds__(256) void vq_e2(const float* __restrict__ emb,
                                             float* __restrict__ e2) {
    int k = blockIdx.x * 256 + threadIdx.x;
    if (k < K_DIM) {
        float s = 0.f;
        for (int d = 0; d < D_DIM; ++d) {
            float v = emb[(size_t)d * K_DIM + k];
            s = fmaf(v, v, s);
        }
        e2[k] = s;
    }
}

// Prep 3: transposed fp32 codebook embT[k][d] for coalesced output gather
__global__ __launch_bounds__(256) void vq_transpose(const float* __restrict__ emb,
                                                    float* __restrict__ embT) {
    int t = blockIdx.x * 256 + threadIdx.x;       // 256*2048 elements
    int d = t >> 11;
    int k = t & 2047;
    embT[(size_t)k * D_DIM + d] = emb[t];
}

__device__ __forceinline__ void acc_sq(float& s, const float4& q, const float4& xv) {
    s = fmaf(q.x - xv.x, q.x - xv.x, s);
    s = fmaf(q.y - xv.y, q.y - xv.y, s);
    s = fmaf(q.z - xv.z, q.z - xv.z, s);
    s = fmaf(q.w - xv.w, q.w - xv.w, s);
}

// Async-stage one 32KB slab (256 threads x 8 x b128) global -> LDS, ASYNCcnt.
__device__ __forceinline__ void async_slab(const _Float16* src, unsigned ldsDst,
                                           int tid) {
    unsigned long long sb = (unsigned long long)src;
    unsigned off = (unsigned)tid * 16u;
#pragma unroll
    for (int i = 0; i < 8; ++i) {
        asm volatile("global_load_async_to_lds_b128 %0, %1, %2"
                     :: "v"(ldsDst + off + (unsigned)i * 4096u),
                        "v"(off + (unsigned)i * 4096u),
                        "s"(sb)
                     : "memory");
    }
}

// ---------------------------------------------------------------------------
// Main kernel: double-buffered async LDS staging + f16 WMMA distance sweep
// ---------------------------------------------------------------------------
__global__ __launch_bounds__(WG_THREADS) void vq_main(
    const float* __restrict__ x, const _Float16* __restrict__ embP,
    const float* __restrict__ e2, const float* __restrict__ embT,
    float* __restrict__ out, float* __restrict__ partials) {

    __shared__ v16h ldsB[2 * SLAB_V16];          // 64 KB, double buffered

    const int tid  = threadIdx.x;
    const int lane = tid & 31;
    const int w    = tid >> 5;
    const int rowBase = blockIdx.x * ROWS_WG + w * 16;
    const int row     = rowBase + (lane & 15);
    const int kbase   = (lane < 16) ? 0 : 16;    // contiguous k = 16h + s (matches B pack)
    const int colOff  = lane & 15;

    const unsigned ldsBase = (unsigned)(uintptr_t)&ldsB[0];

    // prologue: kick off async stage of group 0 into buffer 0
    async_slab(embP, ldsBase, tid);

    // ---- load A fragments once (f32 -> f16); overlaps with the async copy
    v16h aF[N_CHUNKS];
    const float* xrow = x + (size_t)row * D_DIM;
#pragma unroll
    for (int c = 0; c < N_CHUNKS; ++c) {
        const float* p = xrow + 32 * c + kbase;
        float4 f0 = *(const float4*)(p);
        float4 f1 = *(const float4*)(p + 4);
        float4 g0 = *(const float4*)(p + 8);
        float4 g1 = *(const float4*)(p + 12);
        v16h a;
        a[0]=(_Float16)f0.x; a[1]=(_Float16)f0.y; a[2]=(_Float16)f0.z; a[3]=(_Float16)f0.w;
        a[4]=(_Float16)f1.x; a[5]=(_Float16)f1.y; a[6]=(_Float16)f1.z; a[7]=(_Float16)f1.w;
        a[8]=(_Float16)g0.x; a[9]=(_Float16)g0.y; a[10]=(_Float16)g0.z; a[11]=(_Float16)g0.w;
        a[12]=(_Float16)g1.x; a[13]=(_Float16)g1.y; a[14]=(_Float16)g1.z; a[15]=(_Float16)g1.w;
        aF[c] = a;
    }

    float bestVal[8];
    int   bestIdx[8];
#pragma unroll
    for (int r = 0; r < 8; ++r) { bestVal[r] = 3.4e38f; bestIdx[r] = 0; }

    for (int g = 0; g < N_GROUPS; ++g) {
        const int curBuf = (g & 1) * SLAB_V16;

        // issue next slab into the other buffer, then wait for current slab:
        // async loads complete in order, so <=8 outstanding => current done.
        if (g + 1 < N_GROUPS) {
            async_slab(embP + (size_t)(g + 1) * (SLAB_BYTES / 2),
                       ldsBase + (unsigned)(((g + 1) & 1) * SLAB_BYTES), tid);
            asm volatile("s_wait_asynccnt 0x8" ::: "memory");
        } else {
            asm volatile("s_wait_asynccnt 0x0" ::: "memory");
        }

        // hoist e2 loads; latency overlaps barrier + WMMA bursts
        float e2g[TG];
#pragma unroll
        for (int tl = 0; tl < TG; ++tl)
            e2g[tl] = e2[(g * TG + tl) * 16 + colOff];

        __syncthreads();   // publish current slab (others' async writes)

        for (int tl = 0; tl < TG; ++tl) {
            v16h bF[N_CHUNKS];
#pragma unroll
            for (int c = 0; c < N_CHUNKS; ++c)
                bF[c] = ldsB[curBuf + (tl * N_CHUNKS + c) * 32 + lane];

            // two independent accumulator chains -> alternating WMMAs pipeline
            v8f acc0 = {}, acc1 = {};
#pragma unroll
            for (int c = 0; c < N_CHUNKS; c += 2) {
                acc0 = __builtin_amdgcn_wmma_f32_16x16x32_f16(
                    false, aF[c], false, bF[c], (short)0, acc0, false, false);
                acc1 = __builtin_amdgcn_wmma_f32_16x16x32_f16(
                    false, aF[c + 1], false, bF[c + 1], (short)0, acc1, false, false);
            }

            const int col = (g * TG + tl) * 16 + colOff;
            const float e2c = e2g[tl];
#pragma unroll
            for (int r = 0; r < 8; ++r) {
                float s = fmaf(-2.f, acc0[r] + acc1[r], e2c); // ||e||^2 - 2 f.e
                if (s < bestVal[r]) { bestVal[r] = s; bestIdx[r] = col; } // first-min
            }
        }
        __syncthreads();   // done reading curBuf; next iter may refill other buf
    }

    // ---- argmin across the 16 lanes of each half-wave (rows r / 8+r)
#pragma unroll
    for (int r = 0; r < 8; ++r) {
        float v = bestVal[r];
        int   i = bestIdx[r];
#pragma unroll
        for (int m = 1; m < 16; m <<= 1) {
            float v2 = __shfl_xor(v, m, 32);
            int   i2 = __shfl_xor(i, m, 32);
            if (v2 < v || (v2 == v && i2 < i)) { v = v2; i = i2; }
        }
        bestIdx[r] = i;
    }

    // ---- coalesced gather from embT, write out = q, accumulate (q-x)^2
    float lsum = 0.f;
#pragma unroll
    for (int r = 0; r < 8; ++r) {
        const int idxA = __shfl(bestIdx[r], 0, 32);   // row rowBase + r
        const int idxB = __shfl(bestIdx[r], 16, 32);  // row rowBase + 8 + r
        const size_t row0 = (size_t)(rowBase + r) * D_DIM;
        const size_t row1 = (size_t)(rowBase + 8 + r) * D_DIM;
        const float4* qA = (const float4*)(embT + (size_t)idxA * D_DIM);
        const float4* qB = (const float4*)(embT + (size_t)idxB * D_DIM);
        const float4* xA = (const float4*)(x + row0);
        const float4* xB = (const float4*)(x + row1);
        float4* oA = (float4*)(out + row0);
        float4* oB = (float4*)(out + row1);
#pragma unroll
        for (int j = 0; j < 2; ++j) {
            const int d4 = lane + 32 * j;             // float4 index 0..63
            float4 qa = qA[d4], xa = xA[d4];
            float4 qb = qB[d4], xb = xB[d4];
            oA[d4] = qa;
            oB[d4] = qb;
            acc_sq(lsum, qa, xa);
            acc_sq(lsum, qb, xb);
        }
    }
#pragma unroll
    for (int m = 1; m < 32; m <<= 1) lsum += __shfl_xor(lsum, m, 32);

    // per-wave partials through (reused) LDS
    float* wsum = (float*)ldsB;
    if (lane == 0) wsum[w] = lsum;
    __syncthreads();
    if (tid == 0) {
        float t = 0.f;
        for (int i = 0; i < WAVES_WG; ++i) t += wsum[i];
        partials[blockIdx.x] = t;
    }
}

// Deterministic fixed-order final reduction -> vq_loss scalar
__global__ __launch_bounds__(256) void vq_finalize(const float* __restrict__ partials,
                                                   float* __restrict__ loss) {
    __shared__ float s[256];
    int tid = threadIdx.x;
    float t = 0.f;
    for (int i = tid; i < N_WG; i += 256) t += partials[i];
    s[tid] = t;
    __syncthreads();
    for (int k = 128; k > 0; k >>= 1) {
        if (tid < k) s[tid] += s[tid + k];
        __syncthreads();
    }
    if (tid == 0) loss[0] = -0.75f * s[0] / ND_TOTAL;  // (beta-1)*MSE
}

extern "C" void kernel_launch(void* const* d_in, const int* in_sizes, int n_in,
                              void* d_out, int out_size, void* d_ws, size_t ws_size,
                              hipStream_t stream) {
    const float* x   = (const float*)d_in[0];   // [64,32,32,256]
    const float* emb = (const float*)d_in[1];   // [256,2048]
    float* out = (float*)d_out;                 // 16777216 outs + 1 loss

    _Float16* embP    = (_Float16*)d_ws;                                  // 1 MB
    float*    e2      = (float*)((char*)d_ws + (size_t)K_DIM * D_DIM * 2);// 8 KB
    float*    partial = e2 + K_DIM;                                       // 2 KB
    float*    embT    = (float*)((char*)d_ws + (1u << 20) + 16384);       // 2 MB

    vq_pack_b<<<128, 256, 0, stream>>>(emb, embP);
    vq_e2<<<K_DIM / 256, 256, 0, stream>>>(emb, e2);
    vq_transpose<<<(D_DIM * K_DIM) / 256, 256, 0, stream>>>(emb, embT);
    vq_main<<<N_WG, WG_THREADS, 0, stream>>>(x, embP, e2, embT, out, partial);
    vq_finalize<<<1, 256, 0, stream>>>(partial, out + (size_t)N_ROWS * D_DIM);
}
